// OmniPathGuidedGPSLayer_17334488006966
// MI455X (gfx1250) — matrix operand order repaired and checked by reading, not compiled, not run on hardware
//
#include <hip/hip_runtime.h>
#include <hip/hip_bf16.h>

typedef __attribute__((ext_vector_type(16))) __bf16 v16bf;
typedef __attribute__((ext_vector_type(8)))  float  v8f;
typedef unsigned int v4u __attribute__((ext_vector_type(4)));
typedef int v4i __attribute__((ext_vector_type(4)));
typedef int v8i __attribute__((ext_vector_type(8)));

#define N_NODES 4096
#define N_EDGES 262144
#define HDIM    256
#define EDIM    6
#define NTYPE   4
#define NHEAD   8
#define HD      32   // head dim

// ---------------------------------------------------------------- utilities

__global__ void zero_kernel(float* p, int n) {
    int i = blockIdx.x * blockDim.x + threadIdx.x;
    if (i < n) p[i] = 0.0f;
}

__global__ void softmax_w_kernel(const float* __restrict__ etw, float* __restrict__ wsm) {
    if (threadIdx.x == 0 && blockIdx.x == 0) {
        float m = etw[0];
        for (int t = 1; t < NTYPE; ++t) m = fmaxf(m, etw[t]);
        float e[NTYPE]; float s = 0.f;
        for (int t = 0; t < NTYPE; ++t) { e[t] = __expf(etw[t] - m); s += e[t]; }
        for (int t = 0; t < NTYPE; ++t) wsm[t] = e[t] / s;
    }
}

__global__ void add_kernel(const float* __restrict__ a, const float* __restrict__ b,
                           float* __restrict__ c, int n) {
    int i = blockIdx.x * blockDim.x + threadIdx.x;
    if (i < n) c[i] = a[i] + b[i];
}

__global__ void concat_kernel(const float* __restrict__ a, const float* __restrict__ b,
                              float* __restrict__ z) {
    int i = blockIdx.x * blockDim.x + threadIdx.x;   // over N * 2H
    if (i >= N_NODES * 2 * HDIM) return;
    int r = i / (2 * HDIM), c = i % (2 * HDIM);
    z[i] = (c < HDIM) ? a[r * HDIM + c] : b[r * HDIM + (c - HDIM)];
}

// ---------------------------------------------- GINE edge messages + scatter
__global__ void gine_edge_kernel(const float* __restrict__ x,
                                 const int*   __restrict__ ei,
                                 const float* __restrict__ eattr,
                                 const int*   __restrict__ etype,
                                 const float* __restrict__ We,   // [T][ED][H]
                                 const float* __restrict__ be,   // [T][H]
                                 float* __restrict__ agg) {      // [T][N][H]
    int wave = blockIdx.x * (blockDim.x >> 5) + (threadIdx.x >> 5);
    if (wave >= N_EDGES) return;
    int lane = threadIdx.x & 31;
    int e = wave;
    int s = ei[e];
    int d = ei[N_EDGES + e];
    int t = etype[e];
    float ea[EDIM];
#pragma unroll
    for (int k = 0; k < EDIM; ++k) ea[k] = eattr[e * EDIM + k];
    const float* Wt = We + (size_t)t * EDIM * HDIM;
    const float* bt = be + (size_t)t * HDIM;
    float* at = agg + (size_t)t * N_NODES * HDIM + (size_t)d * HDIM;
#pragma unroll
    for (int j = 0; j < 8; ++j) {
        int c = lane * 8 + j;
        float v = x[(size_t)s * HDIM + c] + bt[c];
#pragma unroll
        for (int k = 0; k < EDIM; ++k) v += ea[k] * Wt[k * HDIM + c];
        v = fmaxf(v, 0.0f);
        atomicAdd(&at[c], v);
    }
}

// ------------------------------------------------ TDM 2D tile load (f32)
// Builds a D# (ISA ch.8) for a 2D f32 tile with LDS row padding and issues
// TENSOR_LOAD_TO_LDS. This toolchain uses the 6-arg builtin form:
// (v4u group0, v8i group1, v4i group2, v4i group3, v8i extra, i32 cpol).
__device__ __forceinline__ void tdm_load_tile_f32(
    void* lds_dst, const float* gsrc,
    unsigned tensor_w, unsigned tensor_h, unsigned long long row_stride_elems,
    unsigned tile_w, unsigned tile_h,
    unsigned pad_interval_code, unsigned pad_amount_code)
{
    unsigned long long ga = (unsigned long long)(uintptr_t)gsrc;
    unsigned lds = (unsigned)(uintptr_t)lds_dst;   // low 32 bits = LDS offset
    v4u g0;
    g0[0] = 1u;                                          // count=1 (valid D#)
    g0[1] = lds;                                         // lds_addr
    g0[2] = (unsigned)(ga & 0xFFFFFFFFu);                // global_addr lo
    g0[3] = (unsigned)((ga >> 32) & 0x01FFFFFFu) | (2u << 30);  // hi | type=2
    v8i g1;
    // d0: wg_mask=0 | data_size=2(4B)<<16 | pad_en<<20 | interval<<22 | amount<<25
    g1[0] = (int)((2u << 16) | (1u << 20) | (pad_interval_code << 22) |
                  (pad_amount_code << 25));
    g1[1] = (int)((tensor_w & 0xFFFFu) << 16);                       // dim0 lo16
    g1[2] = (int)(((tensor_w >> 16) & 0xFFFFu) | ((tensor_h & 0xFFFFu) << 16));
    g1[3] = (int)(((tensor_h >> 16) & 0xFFFFu) | (tile_w << 16));    // tile_dim0
    g1[4] = (int)(tile_h & 0xFFFFu);                                 // tile_dim1
    g1[5] = (int)(row_stride_elems & 0xFFFFFFFFull);                 // stride lo
    g1[6] = (int)((row_stride_elems >> 32) & 0xFFFFull);             // stride hi
    g1[7] = 0;
    v4i z4 = {0, 0, 0, 0};
    v8i z8 = {0, 0, 0, 0, 0, 0, 0, 0};
    __builtin_amdgcn_tensor_load_to_lds(g0, g1, z4, z4, z8, 0);
}

// --------------------------------------------------- blocked bf16 WMMA GEMM
// 256 threads = 8 waves compute a 128x64 C tile. Per K-step of 32, A(128x32)
// and B(32x64) f32 tiles are staged into LDS by the Tensor Data Mover with
// double buffering (TENSORcnt-gated). Each wave computes a 32x32 sub-tile
// with 4 WMMA accumulators. Epilogue: +bias, relu?, *alpha (device scalar),
// optional accumulate.
#define ASTR 34   // LDS row stride for A tile (32 + 2 pad dwords)
#define BSTR 66   // LDS row stride for B tile (64 + 2 pad dwords)

__global__ void gemm_bf16_wmma(const float* __restrict__ A, const float* __restrict__ B,
                               const float* __restrict__ bias, float* __restrict__ C,
                               int M, int N, int K, int relu, int acc,
                               const float* __restrict__ alpha_ptr, int alpha_idx) {
    __shared__ float lA[2][128 * ASTR];
    __shared__ float lB[2][32 * BSTR];

    int tiles_x = N >> 6;                 // N / 64
    int bx = blockIdx.x % tiles_x;
    int by = blockIdx.x / tiles_x;
    int w    = threadIdx.x >> 5;
    int lane = threadIdx.x & 31;
    int half = lane >> 4;
    int l16  = lane & 15;
    int wm = w >> 1;                      // 0..3 -> row group of 32
    int wn = w & 1;                       // 0..1 -> col group of 32
    int rA = wm * 32;
    int cB = wn * 32;

    v8f c00 = {0.f,0.f,0.f,0.f,0.f,0.f,0.f,0.f};
    v8f c01 = c00, c10 = c00, c11 = c00;

    int nk = K >> 5;
    // prologue: stage K-step 0 into buffer 0 (wave 0 issues the TDM ops)
    if (w == 0) {
        tdm_load_tile_f32(&lA[0][0], A + (size_t)(by * 128) * K,
                          (unsigned)K, (unsigned)M, (unsigned)K, 32u, 128u,
                          /*interval 32dw*/4u, /*pad 2dw*/1u);
        tdm_load_tile_f32(&lB[0][0], B + (size_t)(bx * 64),
                          (unsigned)N, (unsigned)K, (unsigned)N, 64u, 32u,
                          /*interval 64dw*/5u, /*pad 2dw*/1u);
    }
    for (int kb = 0; kb < nk; ++kb) {
        int cur = kb & 1;
        if (w == 0) {
            if (kb + 1 < nk) {
                int k1 = (kb + 1) << 5;
                tdm_load_tile_f32(&lA[cur ^ 1][0], A + (size_t)(by * 128) * K + k1,
                                  (unsigned)K, (unsigned)M, (unsigned)K, 32u, 128u,
                                  4u, 1u);
                tdm_load_tile_f32(&lB[cur ^ 1][0], B + (size_t)k1 * N + bx * 64,
                                  (unsigned)N, (unsigned)K, (unsigned)N, 64u, 32u,
                                  5u, 1u);
                __builtin_amdgcn_s_wait_tensorcnt((short)2);   // cur buffer done
            } else {
                __builtin_amdgcn_s_wait_tensorcnt((short)0);
            }
        }
        __syncthreads();   // publish cur buffer to all waves
        // fragments from LDS (f32 -> bf16 on read)
        v16bf a0, a1;
        const float* pa0 = &lA[cur][(rA + l16) * ASTR + half * 8];
        const float* pa1 = &lA[cur][(rA + 16 + l16) * ASTR + half * 8];
#pragma unroll
        for (int i = 0; i < 8; ++i) { a0[i] = (__bf16)pa0[i];      a1[i] = (__bf16)pa1[i]; }
#pragma unroll
        for (int i = 0; i < 8; ++i) { a0[8+i] = (__bf16)pa0[16+i]; a1[8+i] = (__bf16)pa1[16+i]; }
        v16bf b0, b1;
        const float* pb = &lB[cur][(half * 8) * BSTR + cB + l16];
#pragma unroll
        for (int i = 0; i < 16; ++i) { b0[i] = (__bf16)pb[(i + (i & 8)) * BSTR]; b1[i] = (__bf16)pb[(i + (i & 8)) * BSTR + 16]; }

        c00 = __builtin_amdgcn_wmma_f32_16x16x32_bf16(false, a0, false, b0, (short)0, c00, false, false);
        c01 = __builtin_amdgcn_wmma_f32_16x16x32_bf16(false, a0, false, b1, (short)0, c01, false, false);
        c10 = __builtin_amdgcn_wmma_f32_16x16x32_bf16(false, a1, false, b0, (short)0, c10, false, false);
        c11 = __builtin_amdgcn_wmma_f32_16x16x32_bf16(false, a1, false, b1, (short)0, c11, false, false);
        __syncthreads();   // all reads of cur done before it is overwritten
    }

    float alpha = alpha_ptr ? alpha_ptr[alpha_idx] : 1.0f;
    int col0 = bx * 64 + cB + l16;
    float bv0 = bias ? bias[col0] : 0.0f;
    float bv1 = bias ? bias[col0 + 16] : 0.0f;
#pragma unroll
    for (int i = 0; i < 8; ++i) {
        int row0 = by * 128 + rA + i + 8 * half;
        int row1 = row0 + 16;
        float v;
        float* p;
        v = c00[i] + bv0; if (relu) v = fmaxf(v, 0.f); v *= alpha;
        p = C + (size_t)row0 * N + col0;      if (acc) *p += v; else *p = v;
        v = c01[i] + bv1; if (relu) v = fmaxf(v, 0.f); v *= alpha;
        p = C + (size_t)row0 * N + col0 + 16; if (acc) *p += v; else *p = v;
        v = c10[i] + bv0; if (relu) v = fmaxf(v, 0.f); v *= alpha;
        p = C + (size_t)row1 * N + col0;      if (acc) *p += v; else *p = v;
        v = c11[i] + bv1; if (relu) v = fmaxf(v, 0.f); v *= alpha;
        p = C + (size_t)row1 * N + col0 + 16; if (acc) *p += v; else *p = v;
    }
}

// ------------------------------------------------------- flash attention
// one wave = 16 query rows of one head (hd=32). Streams 32-key blocks.
__global__ void flash_attn_kernel(const float* __restrict__ Q,
                                  const float* __restrict__ Km,
                                  const float* __restrict__ Vm,
                                  const float* __restrict__ bias,
                                  float* __restrict__ O) {
    __shared__ __bf16 Plds[8][16][32];
    int w = threadIdx.x >> 5;
    int wid = blockIdx.x * 8 + w;           // 0 .. 2047
    int h  = wid >> 8;                       // head
    int rb = wid & 255;                      // row block
    int m0 = rb * 16;
    int lane = threadIdx.x & 31;
    int half = lane >> 4;
    int l16  = lane & 15;
    const float scl = 0.17677669529663687f;  // 1/sqrt(32)

    v16bf qa;
    const float* qp = Q + (size_t)(m0 + l16) * HDIM + h * HD + half * 8;
#pragma unroll
    for (int i = 0; i < 8; ++i) qa[i]     = (__bf16)qp[i];
#pragma unroll
    for (int i = 0; i < 8; ++i) qa[8 + i] = (__bf16)qp[16 + i];

    v8f o0 = {0.f,0.f,0.f,0.f,0.f,0.f,0.f,0.f};
    v8f o1 = o0;
    float rmax[8], rsum[8];
#pragma unroll
    for (int i = 0; i < 8; ++i) { rmax[i] = -1e30f; rsum[i] = 0.f; }

    for (int j0 = 0; j0 < N_NODES; j0 += 32) {
        float s0[8], s1[8];
#pragma unroll
        for (int jt = 0; jt < 2; ++jt) {
            v16bf kb;
            const float* kp = Km + (size_t)(j0 + jt * 16 + l16) * HDIM + h * HD + half * 8;
#pragma unroll
            for (int i = 0; i < 16; ++i) kb[i] = (__bf16)kp[i + (i & 8)];
            v8f sc = {0.f,0.f,0.f,0.f,0.f,0.f,0.f,0.f};
            sc = __builtin_amdgcn_wmma_f32_16x16x32_bf16(false, qa, false, kb,
                                                         (short)0, sc, false, false);
            int jcol = j0 + jt * 16 + l16;
#pragma unroll
            for (int i = 0; i < 8; ++i) {
                int row = m0 + i + 8 * half;
                float v = sc[i] * scl + bias[(size_t)row * N_NODES + jcol];
                if (jt == 0) s0[i] = v; else s1[i] = v;
            }
        }
#pragma unroll
        for (int i = 0; i < 8; ++i) {
            float bm = fmaxf(s0[i], s1[i]);
            bm = fmaxf(bm, __shfl_xor(bm, 1, 32));
            bm = fmaxf(bm, __shfl_xor(bm, 2, 32));
            bm = fmaxf(bm, __shfl_xor(bm, 4, 32));
            bm = fmaxf(bm, __shfl_xor(bm, 8, 32));
            float mnew = fmaxf(rmax[i], bm);
            float corr = __expf(rmax[i] - mnew);
            rmax[i] = mnew;
            float p0 = __expf(s0[i] - mnew);
            float p1 = __expf(s1[i] - mnew);
            float ps = p0 + p1;
            ps += __shfl_xor(ps, 1, 32);
            ps += __shfl_xor(ps, 2, 32);
            ps += __shfl_xor(ps, 4, 32);
            ps += __shfl_xor(ps, 8, 32);
            rsum[i] = rsum[i] * corr + ps;
            o0[i] *= corr;
            o1[i] *= corr;
            Plds[w][i + 8 * half][l16]      = (__bf16)p0;
            Plds[w][i + 8 * half][16 + l16] = (__bf16)p1;
        }
        asm volatile("s_wait_dscnt 0x0" ::: "memory");
        v16bf pa;
#pragma unroll
        for (int i = 0; i < 8; ++i) pa[i]     = Plds[w][l16][half * 8 + i];
#pragma unroll
        for (int i = 0; i < 8; ++i) pa[8 + i] = Plds[w][l16][half * 8 + 16 + i];
#pragma unroll
        for (int nt = 0; nt < 2; ++nt) {
            v16bf vb;
            const float* vp = Vm + (size_t)(j0 + half * 8) * HDIM + h * HD + nt * 16 + l16;
#pragma unroll
            for (int i = 0; i < 16; ++i) vb[i] = (__bf16)vp[(size_t)(i + (i & 8)) * HDIM];
            if (nt == 0)
                o0 = __builtin_amdgcn_wmma_f32_16x16x32_bf16(false, pa, false, vb,
                                                             (short)0, o0, false, false);
            else
                o1 = __builtin_amdgcn_wmma_f32_16x16x32_bf16(false, pa, false, vb,
                                                             (short)0, o1, false, false);
        }
    }
#pragma unroll
    for (int i = 0; i < 8; ++i) {
        float inv = 1.0f / rsum[i];
        int row = m0 + i + 8 * half;
        O[(size_t)row * HDIM + h * HD + l16]      = o0[i] * inv;
        O[(size_t)row * HDIM + h * HD + 16 + l16] = o1[i] * inv;
    }
}

// ------------------------------------------------------------- batch norm
__global__ void bn_stats_kernel(const float* __restrict__ z, float* __restrict__ stats) {
    __shared__ float as[256], bs[256];
    int ch = blockIdx.x, tid = threadIdx.x;
    float s = 0.f, ss = 0.f;
    for (int r = tid; r < N_NODES; r += 256) {
        float v = z[(size_t)r * HDIM + ch];
        s += v; ss += v * v;
    }
    as[tid] = s; bs[tid] = ss;
    __syncthreads();
    for (int st = 128; st > 0; st >>= 1) {
        if (tid < st) { as[tid] += as[tid + st]; bs[tid] += bs[tid + st]; }
        __syncthreads();
    }
    if (tid == 0) { stats[ch] = as[0]; stats[HDIM + ch] = bs[0]; }
}

__global__ void bn_apply_kernel(const float* __restrict__ x, const float* __restrict__ z,
                                const float* __restrict__ stats,
                                const float* __restrict__ gamma, const float* __restrict__ beta,
                                float* __restrict__ out) {
    int i = blockIdx.x * blockDim.x + threadIdx.x;
    if (i >= N_NODES * HDIM) return;
    int c = i % HDIM;
    float mu  = stats[c] / (float)N_NODES;
    float var = stats[HDIM + c] / (float)N_NODES - mu * mu;
    float zn  = (z[i] - mu) * rsqrtf(var + 1e-5f);
    out[i] = x[i] + gamma[c] * zn + beta[c];
}

// ------------------------------------------------------------------ launch
static inline void launch_gemm(const float* A, const float* B, const float* bias, float* C,
                               int M, int N, int K, int relu, int acc,
                               const float* aptr, int aidx, hipStream_t s) {
    int blocks = (M / 128) * (N / 64);
    gemm_bf16_wmma<<<blocks, 256, 0, s>>>(A, B, bias, C, M, N, K, relu, acc, aptr, aidx);
}

extern "C" void kernel_launch(void* const* d_in, const int* in_sizes, int n_in,
                              void* d_out, int out_size, void* d_ws, size_t ws_size,
                              hipStream_t stream) {
    const float* x      = (const float*)d_in[0];
    const int*   ei     = (const int*)  d_in[1];
    const float* eattr  = (const float*)d_in[2];
    const int*   etype  = (const int*)  d_in[3];
    const float* biasM  = (const float*)d_in[4];
    const float* We     = (const float*)d_in[5];
    const float* be     = (const float*)d_in[6];
    const float* W1     = (const float*)d_in[7];
    const float* b1     = (const float*)d_in[8];
    const float* W2     = (const float*)d_in[9];
    const float* b2     = (const float*)d_in[10];
    const float* etw    = (const float*)d_in[11];
    const float* Wq     = (const float*)d_in[12];
    const float* bq     = (const float*)d_in[13];
    const float* Wk     = (const float*)d_in[14];
    const float* bk     = (const float*)d_in[15];
    const float* Wv     = (const float*)d_in[16];
    const float* bv     = (const float*)d_in[17];
    const float* Wo     = (const float*)d_in[18];
    const float* bo     = (const float*)d_in[19];
    const float* fW1    = (const float*)d_in[20];
    const float* fb1    = (const float*)d_in[21];
    const float* fW2    = (const float*)d_in[22];
    const float* fb2    = (const float*)d_in[23];
    const float* gamma  = (const float*)d_in[24];
    const float* beta   = (const float*)d_in[25];
    float* out = (float*)d_out;

    const size_t NH = (size_t)N_NODES * HDIM;   // 1,048,576
    float* ws = (float*)d_ws;
    float* agg   = ws;                 // 4*NH
    float* mpnn  = ws + 4 * NH;        // NH (accumulated)
    float* stats = ws + 5 * NH;        // 512
    float* wsm   = ws + 5 * NH + 512;  // 4
    float* ht    = ws + 5 * NH + 1024; // NH
    float* tmp1  = ht + NH;            // 2*NH
    float* Qb    = tmp1 + 2 * NH;      // NH
    float* Kb    = Qb + NH;            // NH
    float* Vb    = Kb + NH;            // NH
    float* ah    = Vb + NH;            // NH
    float* ao    = ah + NH;            // NH
    float* zc    = ao + NH;            // 2*NH
    float* fh    = zc + 2 * NH;        // 4*NH
    float* z2    = fh + 4 * NH;        // NH

    {
        int n = (int)(5 * NH + 512);
        zero_kernel<<<(n + 255) / 256, 256, 0, stream>>>(ws, n);
    }
    softmax_w_kernel<<<1, 32, 0, stream>>>(etw, wsm);
    gine_edge_kernel<<<N_EDGES / 8, 256, 0, stream>>>(x, ei, eattr, etype, We, be, agg);
    for (int t = 0; t < NTYPE; ++t) {
        add_kernel<<<(int)(NH / 256), 256, 0, stream>>>(x, agg + (size_t)t * NH, ht, (int)NH);
        launch_gemm(ht, W1 + (size_t)t * HDIM * 2 * HDIM, b1 + (size_t)t * 2 * HDIM, tmp1,
                    N_NODES, 2 * HDIM, HDIM, /*relu*/1, /*acc*/0, nullptr, 0, stream);
        launch_gemm(tmp1, W2 + (size_t)t * 2 * HDIM * HDIM, b2 + (size_t)t * HDIM, mpnn,
                    N_NODES, HDIM, 2 * HDIM, /*relu*/0, /*acc*/1, wsm, t, stream);
    }
    launch_gemm(x, Wq, bq, Qb, N_NODES, HDIM, HDIM, 0, 0, nullptr, 0, stream);
    launch_gemm(x, Wk, bk, Kb, N_NODES, HDIM, HDIM, 0, 0, nullptr, 0, stream);
    launch_gemm(x, Wv, bv, Vb, N_NODES, HDIM, HDIM, 0, 0, nullptr, 0, stream);
    flash_attn_kernel<<<256, 256, 0, stream>>>(Qb, Kb, Vb, biasM, ah);
    launch_gemm(ah, Wo, bo, ao, N_NODES, HDIM, HDIM, 0, 0, nullptr, 0, stream);
    concat_kernel<<<(int)(2 * NH / 256), 256, 0, stream>>>(mpnn, ao, zc);
    launch_gemm(zc, fW1, fb1, fh, N_NODES, 4 * HDIM, 2 * HDIM, /*relu*/1, 0, nullptr, 0, stream);
    launch_gemm(fh, fW2, fb2, z2, N_NODES, HDIM, 4 * HDIM, 0, 0, nullptr, 0, stream);
    bn_stats_kernel<<<HDIM, 256, 0, stream>>>(z2, stats);
    bn_apply_kernel<<<(int)(NH / 256), 256, 0, stream>>>(x, z2, stats, gamma, beta, out);
}
